// MaskedAttention_84576495993421
// MI455X (gfx1250) — matrix operand order repaired and checked by reading, not compiled, not run on hardware
//
#include <hip/hip_runtime.h>

// Problem constants (match reference)
constexpr int Bb  = 4;
constexpr int Ss  = 2048;
constexpr int Ee  = 1024;
constexpr int Hh  = 16;
constexpr int Dd  = 64;
constexpr int N3E = 3 * Ee;   // 3072

typedef _Float16 v8h  __attribute__((ext_vector_type(8)));
typedef _Float16 v16h __attribute__((ext_vector_type(16)));
typedef float    v8f  __attribute__((ext_vector_type(8)));

static __device__ __forceinline__ v16h cat8(v8h lo, v8h hi) {
  return __builtin_shufflevector(lo, hi, 0,1,2,3,4,5,6,7,8,9,10,11,12,13,14,15);
}

// A-matrix fragment (16x32 f16, M = lane%16):
// halves 0..7  -> K = 8*(lane/16) + 0..7
// halves 8..15 -> K = 16 + 8*(lane/16) + 0..7
static __device__ __forceinline__ v16h load_afrag(const _Float16* p, int half) {
  v8h lo = *(const v8h*)(p + 8 * half);
  v8h hi = *(const v8h*)(p + 8 * half + 16);
  return cat8(lo, hi);
}

// B-matrix fragment (32x16 f16, N = lane%16): halves 0..7 -> K = 8*(lane/16) + 0..7,
// halves 8..15 -> K = 16 + 8*(lane/16) + 0..7 (p = row base of this lane's N column in K-major storage)
static __device__ __forceinline__ v16h load_bfrag(const _Float16* p, int half) {
  v8h lo = *(const v8h*)(p + 8 * half);
  v8h hi = *(const v8h*)(p + 8 * half + 16);
  return cat8(lo, hi);
}

static __device__ __forceinline__ v8f wmma_f16(v16h a, v16h b, v8f c) {
  return __builtin_amdgcn_wmma_f32_16x16x32_f16(false, a, false, b, (short)0, c, false, false);
}

// ---------------- prep: fp32 -> f16 ----------------
__global__ void cvt_f32_to_f16(const float* __restrict__ src,
                               _Float16* __restrict__ dst, int n) {
  int i = blockIdx.x * blockDim.x + threadIdx.x;
  if (i < n) dst[i] = (_Float16)src[i];
}

// W [E, 3E] fp32 -> Wt [3E, E] f16  (so GEMM B-fragments are contiguous in K=E)
__global__ void transpose_w_f16(const float* __restrict__ W,
                                _Float16* __restrict__ Wt) {
  int i = blockIdx.x * blockDim.x + threadIdx.x;     // over 3E*E
  if (i >= N3E * Ee) return;
  int n = i >> 10;          // row of Wt (0..3071)
  int e = i & (Ee - 1);     // col of Wt (0..1023)
  Wt[i] = (_Float16)W[e * N3E + n];
}

// ---------------- kernel 1: fused QKV GEMM ----------------
// C[8192,3072] = Xh[8192,1024] * Wt^T ; outputs scattered to Q/K/V^T f16 layouts.
__global__ __launch_bounds__(32)
void qkv_gemm(const _Float16* __restrict__ Xh, const _Float16* __restrict__ Wt,
              _Float16* __restrict__ Qh, _Float16* __restrict__ Kh,
              _Float16* __restrict__ Vt) {
  const int lane = threadIdx.x;
  const int col  = lane & 15;
  const int half = lane >> 4;
  const int m0 = blockIdx.y * 64;
  const int n0 = blockIdx.x * 64;

  const _Float16* Ap[4];
  const _Float16* Bp[4];
#pragma unroll
  for (int i = 0; i < 4; ++i) Ap[i] = Xh + (size_t)(m0 + 16 * i + col) * Ee;
#pragma unroll
  for (int j = 0; j < 4; ++j) Bp[j] = Wt + (size_t)(n0 + 16 * j + col) * Ee;

  v8f acc[4][4] = {};

  for (int kk = 0; kk < Ee; kk += 32) {
    if (kk + 32 < Ee) {
#pragma unroll
      for (int j = 0; j < 4; ++j) __builtin_prefetch(Bp[j] + kk + 32, 0, 1);
      __builtin_prefetch(Ap[0] + kk + 32, 0, 1);
    }
    v16h bf[4];
#pragma unroll
    for (int j = 0; j < 4; ++j) bf[j] = load_bfrag(Bp[j] + kk, half);
#pragma unroll
    for (int i = 0; i < 4; ++i) {
      v16h af = load_afrag(Ap[i] + kk, half);
#pragma unroll
      for (int j = 0; j < 4; ++j) acc[i][j] = wmma_f16(af, bf[j], acc[i][j]);
    }
  }

  // C layout: lane holds N = col, rows M = r + 8*half (per VGPR r)
#pragma unroll
  for (int j = 0; j < 4; ++j) {
    const int n = n0 + 16 * j + col;
    const int chunkId = n >> 10;        // 0=Q,1=K,2=V
    const int e = n & (Ee - 1);
    const int h = e >> 6;
    const int d = e & (Dd - 1);
#pragma unroll
    for (int i = 0; i < 4; ++i) {
      const int mbase = m0 + 16 * i + 8 * half;
#pragma unroll
      for (int r = 0; r < 8; ++r) {
        const int m = mbase + r;
        const int b = m >> 11;           // /S
        const int s = m & (Ss - 1);
        const int bh = b * Hh + h;
        const _Float16 v = (_Float16)acc[i][j][r];
        if (chunkId == 0)      Qh[((size_t)bh * Ss + s) * Dd + d] = v;
        else if (chunkId == 1) Kh[((size_t)bh * Ss + s) * Dd + d] = v;
        else                   Vt[((size_t)bh * Dd + d) * Ss + s] = v;
      }
    }
  }
}

// ---------------- kernel 2: causal flash attention ----------------
// One wave handles 16 queries of one (b,h). Computes S^T = K*Q^T so the
// softmax reduction over keys is in-lane + one half-wave shuffle, then
// out^T = V^T * P^T with P re-laid-out through LDS.
__global__ __launch_bounds__(32)
void flash_attn(const _Float16* __restrict__ Qh, const _Float16* __restrict__ Kh,
                const _Float16* __restrict__ Vt, float* __restrict__ out) {
  __shared__ __attribute__((aligned(16))) _Float16 Pq[16][32];

  const int lane = threadIdx.x;
  const int col  = lane & 15;
  const int half = lane >> 4;
  const int q0 = blockIdx.x * 16;
  const int bh = blockIdx.y;
  const int b = bh >> 4;
  const int h = bh & (Hh - 1);

  const _Float16* Qp = Qh + (size_t)bh * Ss * Dd;
  const _Float16* Kp = Kh + (size_t)bh * Ss * Dd;
  const _Float16* Vp = Vt + (size_t)bh * Dd * Ss;

  // Q^T B-fragments (k-dim = d, n = query), reused every key chunk
  v16h qf[2];
#pragma unroll
  for (int dc = 0; dc < 2; ++dc)
    qf[dc] = load_bfrag(Qp + (size_t)(q0 + col) * Dd + 32 * dc, half);

  v8f ot[4] = {};                 // out^T tiles: d rows 16t..16t+15, cols = queries
  float mr = -1e30f, lr = 0.0f;   // online softmax state for query q0+col
  const int q_g = q0 + col;
  const float scale = 0.125f;     // 1/sqrt(64)

  for (int k0 = 0; k0 <= q0 + 15; k0 += 32) {
    // S^T tiles: [16 keys x 16 queries] each
    v8f st[2] = {};
#pragma unroll
    for (int t = 0; t < 2; ++t) {
      const _Float16* krow = Kp + (size_t)(k0 + 16 * t + col) * Dd;
#pragma unroll
      for (int dc = 0; dc < 2; ++dc)
        st[t] = wmma_f16(load_afrag(krow + 32 * dc, half), qf[dc], st[t]);
    }

    // scale + causal mask + chunk max (keys for this lane: r + 8*half + 16*t)
    float cm = -1e30f;
#pragma unroll
    for (int t = 0; t < 2; ++t) {
#pragma unroll
      for (int r = 0; r < 8; ++r) {
        const int key = k0 + 16 * t + 8 * half + r;
        float sv = st[t][r] * scale;
        sv = (key <= q_g) ? sv : -1e30f;
        st[t][r] = sv;
        cm = fmaxf(cm, sv);
      }
    }
    cm = fmaxf(cm, __shfl_xor(cm, 16, 32));   // combine key halves
    const float mn    = fmaxf(mr, cm);
    const float alpha = __expf(mr - mn);

    float rs = 0.0f;
#pragma unroll
    for (int t = 0; t < 2; ++t) {
      v8h ph;
#pragma unroll
      for (int r = 0; r < 8; ++r) {
        const float p = __expf(st[t][r] - mn);
        rs += p;
        ph[r] = (_Float16)p;
      }
      // P stored as [query][key-local]: lane's 8 keys are contiguous
      *(v8h*)&Pq[col][16 * t + 8 * half] = ph;
    }
    rs += __shfl_xor(rs, 16, 32);
    lr = lr * alpha + rs;
    mr = mn;
#pragma unroll
    for (int t = 0; t < 4; ++t) ot[t] *= alpha;

    __syncthreads();   // single-wave block: orders LDS write->read cheaply
    const v16h pf = load_bfrag(&Pq[col][0], half);  // B(k=key, n=query)
#pragma unroll
    for (int t = 0; t < 4; ++t)
      ot[t] = wmma_f16(load_afrag(Vp + (size_t)(16 * t + col) * Ss + k0, half),
                       pf, ot[t]);
    __syncthreads();
  }

  const float inv = 1.0f / lr;
  const size_t base = ((size_t)b * Ss + q_g) * Ee + h * Dd;
#pragma unroll
  for (int t = 0; t < 4; ++t) {
    const int dbase = 16 * t + 8 * half;
    float4 lo4 = make_float4(ot[t][0] * inv, ot[t][1] * inv,
                             ot[t][2] * inv, ot[t][3] * inv);
    float4 hi4 = make_float4(ot[t][4] * inv, ot[t][5] * inv,
                             ot[t][6] * inv, ot[t][7] * inv);
    *(float4*)&out[base + dbase]     = lo4;
    *(float4*)&out[base + dbase + 4] = hi4;
  }
}

// ---------------- host launcher ----------------
extern "C" void kernel_launch(void* const* d_in, const int* in_sizes, int n_in,
                              void* d_out, int out_size, void* d_ws, size_t ws_size,
                              hipStream_t stream) {
  (void)in_sizes; (void)n_in; (void)out_size; (void)ws_size;
  const float* x = (const float*)d_in[0];   // [B,S,E]
  const float* W = (const float*)d_in[1];   // [E,3E]
  float* out = (float*)d_out;               // [B,S,E]

  char* ws = (char*)d_ws;
  _Float16* Xh  = (_Float16*)(ws);                        // 16 MB  (B*S*E)
  _Float16* WhT = (_Float16*)(ws + (16u << 20));          //  6 MB  (3E*E)
  _Float16* Qh  = (_Float16*)(ws + (22u << 20));          // 16 MB  (B*H*S*D)
  _Float16* Kh  = (_Float16*)(ws + (38u << 20));          // 16 MB
  _Float16* Vt  = (_Float16*)(ws + (54u << 20));          // 16 MB  (B*H*D*S)

  const int nx = Bb * Ss * Ee;        // 8,388,608
  const int nw = N3E * Ee;            // 3,145,728

  cvt_f32_to_f16<<<(nx + 255) / 256, 256, 0, stream>>>(x, Xh, nx);
  transpose_w_f16<<<(nw + 255) / 256, 256, 0, stream>>>(W, WhT);

  dim3 ggrid(N3E / 64, (Bb * Ss) / 64);   // (48, 128)
  qkv_gemm<<<ggrid, 32, 0, stream>>>(Xh, WhT, Qh, Kh, Vt);

  dim3 agrid(Ss / 16, Bb * Hh);           // (128, 64)
  flash_attn<<<agrid, 32, 0, stream>>>(Qh, Kh, Vt, out);
}
